// ScaffoldTokenSelector_46024869544428
// MI455X (gfx1250) — matrix operand order repaired and checked by reading, not compiled, not run on hardware
//
#include <hip/hip_runtime.h>

#define BB   4
#define NPTS 16384
#define DIM  768

typedef __attribute__((ext_vector_type(16))) __bf16 v16bf;
typedef __attribute__((ext_vector_type(8)))  __bf16 v8bf;
typedef __attribute__((ext_vector_type(8)))  float  v8f;

__device__ __forceinline__ float sigm(float x) { return 1.f / (1.f + __expf(-x)); }

// ---- WMMA fragment loaders (contiguous k order for A and B, wave32) -------------------
// A (16x32, MxK): lanes 0-15 -> M=lane, K = h                     (h = half idx)
//                 lanes16-31 -> M=lane-16, K = 16 + h
__device__ __forceinline__ v16bf load_a_frag(const __bf16* base, int ld, int k0, int lane) {
  int row = lane & 15;
  int hi  = (lane >> 4) & 1;
  const __bf16* p = base + row * ld + k0 + hi * 16;
  v8bf lo = *(const v8bf*)(p);        // K = k0 + hi*16 + 0..7
  v8bf h8 = *(const v8bf*)(p + 8);    // K = k0 + hi*16 + 8..15
  v16bf r;
#pragma unroll
  for (int i = 0; i < 8; ++i) { r[i] = lo[i]; r[8 + i] = h8[i]; }
  return r;
}
// B (32x16, KxN) from row-major W[n][k] (computing X @ W^T):
// lane L -> N = L%16, K = k0 + (L>=16?16:0) + h  => one contiguous 32B load.
__device__ __forceinline__ v16bf load_b_frag(const __bf16* w, int ld, int k0, int lane, int nvalid) {
  int n  = lane & 15;
  int hi = (lane >> 4) & 1;
  v16bf r;
  if (n < nvalid) {
    r = *(const v16bf*)(w + n * ld + k0 + hi * 16);
  } else {
#pragma unroll
    for (int i = 0; i < 16; ++i) r[i] = (__bf16)0.0f;
  }
  return r;
}

// ---- Kernel 1: farthest point sampling (coords in 192KB LDS, dist in VGPRs) --
__global__ void __launch_bounds__(1024) fps_kernel(const float* __restrict__ coords,
                                                   int* __restrict__ idx_out) {
  extern __shared__ float smem[];                 // 3*NPTS floats (SoA)
  float* sx = smem;
  float* sy = smem + NPTS;
  float* sz = smem + 2 * NPTS;
  __shared__ float s_d[32], s_x[32], s_y[32], s_z[32];
  __shared__ int   s_i[32];
  __shared__ float b_x, b_y, b_z;

  const int b = blockIdx.x;
  const int t = threadIdx.x;
  const float* cb = coords + (size_t)b * NPTS * 3;

  for (int i = t; i < NPTS * 3; i += 1024) {
    int p = i / 3, c = i - p * 3;
    float v = cb[i];
    if (c == 0) sx[p] = v; else if (c == 1) sy[p] = v; else sz[p] = v;
  }
  float dist[16];
#pragma unroll
  for (int j = 0; j < 16; ++j) dist[j] = __builtin_inff();
  __syncthreads();
  if (t == 0) { b_x = sx[0]; b_y = sy[0]; b_z = sz[0]; idx_out[b * 256] = 0; }
  __syncthreads();

  const int lane = t & 31, wave = t >> 5;
  for (int it = 1; it < 256; ++it) {
    float lx = b_x, ly = b_y, lz = b_z;
    float bd = -1.f, bx = 0.f, by = 0.f, bz = 0.f;
    int   bi = 0x7fffffff;
#pragma unroll
    for (int j = 0; j < 16; ++j) {
      int p = t + (j << 10);
      float x = sx[p], y = sy[p], z = sz[p];
      float dx = x - lx, dy = y - ly, dz = z - lz;
      float nd = fminf(dist[j], dx * dx + dy * dy + dz * dz);  // self -> 0, stays 0
      dist[j] = nd;
      if (nd > bd || (nd == bd && p < bi)) { bd = nd; bi = p; bx = x; by = y; bz = z; }
    }
#pragma unroll
    for (int off = 16; off > 0; off >>= 1) {                   // wave32 argmax
      float od = __shfl_xor(bd, off, 32);
      int   oi = __shfl_xor(bi, off, 32);
      float ox = __shfl_xor(bx, off, 32);
      float oy = __shfl_xor(by, off, 32);
      float oz = __shfl_xor(bz, off, 32);
      if (od > bd || (od == bd && oi < bi)) { bd = od; bi = oi; bx = ox; by = oy; bz = oz; }
    }
    if (lane == 0) { s_d[wave] = bd; s_i[wave] = bi; s_x[wave] = bx; s_y[wave] = by; s_z[wave] = bz; }
    __syncthreads();
    if (wave == 0) {
      bd = s_d[lane]; bi = s_i[lane]; bx = s_x[lane]; by = s_y[lane]; bz = s_z[lane];
#pragma unroll
      for (int off = 16; off > 0; off >>= 1) {
        float od = __shfl_xor(bd, off, 32);
        int   oi = __shfl_xor(bi, off, 32);
        float ox = __shfl_xor(bx, off, 32);
        float oy = __shfl_xor(by, off, 32);
        float oz = __shfl_xor(bz, off, 32);
        if (od > bd || (od == bd && oi < bi)) { bd = od; bi = oi; bx = ox; by = oy; bz = oz; }
      }
      if (lane == 0) { b_x = bx; b_y = by; b_z = bz; idx_out[b * 256 + it] = bi; }
    }
    __syncthreads();
  }
}

// ---- Kernel 2: f32 -> bf16 weight conversion --------------------------------
__global__ void conv_bf16_kernel(const float* __restrict__ src, __bf16* __restrict__ dst, int n) {
  int i = blockIdx.x * 256 + threadIdx.x;
  if (i < n) dst[i] = (__bf16)src[i];
}

// ---- Kernel 3: gather center features (bf16) and coords ---------------------
__global__ void gather_kernel(const float* __restrict__ feat, const float* __restrict__ coords,
                              const int* __restrict__ idx, __bf16* __restrict__ cf,
                              float* __restrict__ cc) {
  int b = blockIdx.y, c = blockIdx.x;
  int src = idx[b * 256 + c];
  const float* f = feat + ((size_t)b * NPTS + src) * DIM;
  __bf16* o = cf + ((size_t)b * 256 + c) * DIM;
  for (int k = threadIdx.x; k < DIM; k += 256) o[k] = (__bf16)f[k];
  if (threadIdx.x < 3)
    cc[(b * 256 + c) * 3 + threadIdx.x] = coords[((size_t)b * NPTS + src) * 3 + threadIdx.x];
}

// ---- Kernel 4: the three safety terms ---------------------------------------
__global__ void safety_kernel(const float* __restrict__ cc, float* __restrict__ sg,
                              float* __restrict__ sc, float* __restrict__ sd) {
  __shared__ float r1[256], r2[256];
  int b = blockIdx.x, t = threadIdx.x;
  const float* ccb = cc + b * 256 * 3;

  // global: 1 + 0.95*sigmoid((z-mean)/5) over 256 centers
  float z = ccb[t * 3 + 2];
  r1[t] = z; __syncthreads();
  for (int off = 128; off > 0; off >>= 1) { if (t < off) r1[t] += r1[t + off]; __syncthreads(); }
  float mean = r1[0] * (1.f / 256.f);
  sg[b * 256 + t] = 1.f + 0.95f * sigm((z - mean) * 0.2f);
  __syncthreads();

  // component: 1 + 0.9*exp(-var/0.1), var ddof=1 over first 128 centers
  float zc = (t < 128) ? ccb[t * 3 + 2] : 0.f;
  r1[t] = zc; r2[t] = zc * zc; __syncthreads();
  for (int off = 128; off > 0; off >>= 1) {
    if (t < off) { r1[t] += r1[t + off]; r2[t] += r2[t + off]; }
    __syncthreads();
  }
  float var = (r2[0] - r1[0] * r1[0] * (1.f / 128.f)) * (1.f / 127.f);
  float sv = 1.f + 0.9f * __expf(-var * 10.f);
  if (t < 128) sc[b * 128 + t] = sv;

  // detail: 1 + density/64*0.95 over first 64 centers (d2 < 0.25, self counts)
  if (t < 64) {
    float x = ccb[t * 3], y = ccb[t * 3 + 1], zz = ccb[t * 3 + 2];
    int cnt = 0;
    for (int j = 0; j < 64; ++j) {
      float dx = x - ccb[j * 3], dy = y - ccb[j * 3 + 1], dz = zz - ccb[j * 3 + 2];
      if (dx * dx + dy * dy + dz * dz < 0.25f) cnt++;
    }
    sd[b * 64 + t] = 1.f + (float)cnt * (0.95f / 64.f);
  }
}

// ---- Kernel 5: per-scale scoring: relu(cf@W1^T+b1) -> sigmoid(h@W2^T+b2),
//                score = safety * mean_s(probs).  WMMA bf16, one 16-row M-tile
//                per block, GEMM1 across 8 waves, GEMM2 fused via LDS. --------
__global__ void __launch_bounds__(256) score_kernel(
    const __bf16* __restrict__ cf, const float* __restrict__ safety,
    const __bf16* __restrict__ W1, const float* __restrict__ b1,
    const __bf16* __restrict__ W2, const float* __restrict__ b2,
    float* __restrict__ score, int C, int S) {
  __shared__ __bf16 hLDS[16 * 256];
  int b = blockIdx.y, mt = blockIdx.x;
  int lane = threadIdx.x & 31, wave = threadIdx.x >> 5;
  const __bf16* A = cf + (size_t)b * 256 * DIM + (size_t)mt * 16 * DIM;
  int ntiles = C >> 4;

  for (int nt = wave; nt < ntiles; nt += 8) {
    v8f acc = {0.f, 0.f, 0.f, 0.f, 0.f, 0.f, 0.f, 0.f};
    for (int k0 = 0; k0 < DIM; k0 += 32) {
      v16bf a = load_a_frag(A, DIM, k0, lane);
      v16bf w = load_b_frag(W1 + (size_t)nt * 16 * DIM, DIM, k0, lane, 16);
      acc = __builtin_amdgcn_wmma_f32_16x16x32_bf16(false, a, false, w, (short)0, acc, false, false);
    }
    int n = nt * 16 + (lane & 15);
    float bias = b1[n];
    int mbase = (lane >> 4) << 3;
#pragma unroll
    for (int v = 0; v < 8; ++v)
      hLDS[(mbase + v) * C + n] = (__bf16)fmaxf(acc[v] + bias, 0.f);
  }
  __syncthreads();

  if (wave == 0) {
    v8f acc = {0.f, 0.f, 0.f, 0.f, 0.f, 0.f, 0.f, 0.f};
    for (int k0 = 0; k0 < C; k0 += 32) {
      v16bf a = load_a_frag(hLDS, C, k0, lane);
      v16bf w = load_b_frag(W2, C, k0, lane, S);
      acc = __builtin_amdgcn_wmma_f32_16x16x32_bf16(false, a, false, w, (short)0, acc, false, false);
    }
    int n = lane & 15;
    int mbase = (lane >> 4) << 3;
    float bias = (n < S) ? b2[n] : 0.f;
#pragma unroll
    for (int v = 0; v < 8; ++v) {
      float s = (n < S) ? sigm(acc[v] + bias) : 0.f;   // zero pad cols excluded
      s += __shfl_xor(s, 1, 32);
      s += __shfl_xor(s, 2, 32);
      s += __shfl_xor(s, 4, 32);
      s += __shfl_xor(s, 8, 32);                        // sum within 16-lane half
      if (n == 0) {
        int mg = mt * 16 + mbase + v;
        score[b * C + mg] = safety[b * C + mg] * (s / (float)S);
      }
    }
  }
}

// ---- Kernel 6: top-k per scale (argmax-first ties) + token gather -----------
__global__ void topk_gather_kernel(const float* __restrict__ sg, const float* __restrict__ sc,
                                   const float* __restrict__ sd, const __bf16* __restrict__ cf,
                                   __bf16* __restrict__ tok) {
  __shared__ float s[256];
  __shared__ float rv[256];
  __shared__ int   ri[256];
  __shared__ int   sel[40];
  int b = blockIdx.x, t = threadIdx.x;
  const float* scores[3] = { sg + b * 256, sc + b * 128, sd + b * 64 };
  const int Cs[3] = {256, 128, 64};
  const int Ks[3] = {16, 16, 8};
  int outoff = 0;
  for (int sci = 0; sci < 3; ++sci) {
    int C = Cs[sci], K = Ks[sci];
    if (t < C) s[t] = scores[sci][t];
    __syncthreads();
    for (int k = 0; k < K; ++k) {
      rv[t] = (t < C) ? s[t] : -__builtin_inff();
      ri[t] = t;
      __syncthreads();
      for (int off = 128; off > 0; off >>= 1) {
        if (t < off) {
          if (rv[t + off] > rv[t] || (rv[t + off] == rv[t] && ri[t + off] < ri[t])) {
            rv[t] = rv[t + off]; ri[t] = ri[t + off];
          }
        }
        __syncthreads();
      }
      if (t == 0) { sel[outoff + k] = ri[0]; s[ri[0]] = -__builtin_inff(); }
      __syncthreads();
    }
    outoff += K;
  }
  for (int r = 0; r < 40; ++r) {
    int src = sel[r];
    const __bf16* f = cf + ((size_t)b * 256 + src) * DIM;
    __bf16* o = tok + (size_t)(b * 40 + r) * DIM;
    for (int k = t; k < DIM; k += 256) o[k] = f[k];
  }
}

// ---- Kernel 7: final MLP (768->384 relu ->768) + LayerNorm, WMMA fused ------
__global__ void __launch_bounds__(256) mlp_kernel(
    const __bf16* __restrict__ tok,
    const __bf16* __restrict__ W1, const float* __restrict__ b1,
    const __bf16* __restrict__ W2, const float* __restrict__ b2,
    const float* __restrict__ lng, const float* __restrict__ lnb,
    float* __restrict__ out) {
  __shared__ __bf16 hLDS[16 * 384];   // 12 KB
  __shared__ float  oLDS[16 * 768];   // 48 KB
  int mt = blockIdx.x;
  int lane = threadIdx.x & 31, wave = threadIdx.x >> 5;
  const __bf16* A = tok + (size_t)mt * 16 * DIM;

  for (int nt = wave; nt < 24; nt += 8) {          // N = 384
    v8f acc = {0.f, 0.f, 0.f, 0.f, 0.f, 0.f, 0.f, 0.f};
    for (int k0 = 0; k0 < 768; k0 += 32) {
      v16bf a = load_a_frag(A, 768, k0, lane);
      v16bf w = load_b_frag(W1 + (size_t)nt * 16 * 768, 768, k0, lane, 16);
      acc = __builtin_amdgcn_wmma_f32_16x16x32_bf16(false, a, false, w, (short)0, acc, false, false);
    }
    int n = nt * 16 + (lane & 15);
    float bias = b1[n];
    int mbase = (lane >> 4) << 3;
#pragma unroll
    for (int v = 0; v < 8; ++v)
      hLDS[(mbase + v) * 384 + n] = (__bf16)fmaxf(acc[v] + bias, 0.f);
  }
  __syncthreads();

  for (int nt = wave; nt < 48; nt += 8) {          // N = 768
    v8f acc = {0.f, 0.f, 0.f, 0.f, 0.f, 0.f, 0.f, 0.f};
    for (int k0 = 0; k0 < 384; k0 += 32) {
      v16bf a = load_a_frag(hLDS, 384, k0, lane);
      v16bf w = load_b_frag(W2 + (size_t)nt * 16 * 384, 384, k0, lane, 16);
      acc = __builtin_amdgcn_wmma_f32_16x16x32_bf16(false, a, false, w, (short)0, acc, false, false);
    }
    int n = nt * 16 + (lane & 15);
    float bias = b2[n];
    int mbase = (lane >> 4) << 3;
#pragma unroll
    for (int v = 0; v < 8; ++v)
      oLDS[(mbase + v) * 768 + n] = acc[v] + bias;
  }
  __syncthreads();

  // LayerNorm: row r handled by 16 threads (one 16-lane half of a wave)
  int r = threadIdx.x >> 4, c0 = threadIdx.x & 15;
  float sum = 0.f, sumsq = 0.f;
  for (int j = 0; j < 48; ++j) {
    float v = oLDS[r * 768 + c0 + j * 16];
    sum += v; sumsq += v * v;
  }
  sum   += __shfl_xor(sum, 1, 32);  sum   += __shfl_xor(sum, 2, 32);
  sum   += __shfl_xor(sum, 4, 32);  sum   += __shfl_xor(sum, 8, 32);
  sumsq += __shfl_xor(sumsq, 1, 32); sumsq += __shfl_xor(sumsq, 2, 32);
  sumsq += __shfl_xor(sumsq, 4, 32); sumsq += __shfl_xor(sumsq, 8, 32);
  float mu  = sum * (1.f / 768.f);
  float var = sumsq * (1.f / 768.f) - mu * mu;
  float inv = rsqrtf(var + 1e-5f);
  int rowg = mt * 16 + r;
  for (int j = 0; j < 48; ++j) {
    int col = c0 + j * 16;
    float v = oLDS[r * 768 + col];
    out[(size_t)rowg * 768 + col] = (v - mu) * inv * lng[col] + lnb[col];
  }
}

// ---- Host orchestration -----------------------------------------------------
extern "C" void kernel_launch(void* const* d_in, const int* in_sizes, int n_in,
                              void* d_out, int out_size, void* d_ws, size_t ws_size,
                              hipStream_t stream) {
  (void)in_sizes; (void)n_in; (void)out_size; (void)ws_size;
  const float* feat   = (const float*)d_in[0];
  const float* coords = (const float*)d_in[1];
  const float* gW1 = (const float*)d_in[2];
  const float* gb1 = (const float*)d_in[3];
  const float* gW2 = (const float*)d_in[4];
  const float* gb2 = (const float*)d_in[5];
  const float* cW1 = (const float*)d_in[6];
  const float* cb1 = (const float*)d_in[7];
  const float* cW2 = (const float*)d_in[8];
  const float* cb2 = (const float*)d_in[9];
  const float* dW1 = (const float*)d_in[10];
  const float* db1 = (const float*)d_in[11];
  const float* dW2 = (const float*)d_in[12];
  const float* db2 = (const float*)d_in[13];
  const float* mW1 = (const float*)d_in[14];
  const float* mb1 = (const float*)d_in[15];
  const float* mW2 = (const float*)d_in[16];
  const float* mb2 = (const float*)d_in[17];
  const float* lng = (const float*)d_in[18];
  const float* lnb = (const float*)d_in[19];
  float* out = (float*)d_out;

  char* ws = (char*)d_ws;
  size_t off = 0;
  auto take = [&](size_t bytes) -> char* {
    char* p = ws + off;
    off += (bytes + 255) & ~(size_t)255;
    return p;
  };
  int*    idx  = (int*)take((size_t)BB * 256 * sizeof(int));
  __bf16* cf   = (__bf16*)take((size_t)BB * 256 * DIM * 2);
  float*  cc   = (float*)take((size_t)BB * 256 * 3 * 4);
  float*  sg   = (float*)take((size_t)BB * 256 * 4);
  float*  sc   = (float*)take((size_t)BB * 128 * 4);
  float*  sd   = (float*)take((size_t)BB * 64 * 4);
  float*  scg  = (float*)take((size_t)BB * 256 * 4);
  float*  scc  = (float*)take((size_t)BB * 128 * 4);
  float*  scd  = (float*)take((size_t)BB * 64 * 4);
  __bf16* W1g  = (__bf16*)take(256 * 768 * 2);
  __bf16* W1c  = (__bf16*)take(128 * 768 * 2);
  __bf16* W1d  = (__bf16*)take(64 * 768 * 2);
  __bf16* W2g  = (__bf16*)take(16 * 256 * 2);
  __bf16* W2c  = (__bf16*)take(16 * 128 * 2);
  __bf16* W2d  = (__bf16*)take(8 * 64 * 2);
  __bf16* mW1b = (__bf16*)take(384 * 768 * 2);
  __bf16* mW2b = (__bf16*)take(768 * 384 * 2);
  __bf16* tok  = (__bf16*)take(160 * 768 * 2);

  (void)hipFuncSetAttribute(reinterpret_cast<const void*>(fps_kernel),
                            hipFuncAttributeMaxDynamicSharedMemorySize, 3 * NPTS * 4);

  fps_kernel<<<BB, 1024, 3 * NPTS * 4, stream>>>(coords, idx);

  auto conv = [&](const float* s, __bf16* d, int n) {
    conv_bf16_kernel<<<(n + 255) / 256, 256, 0, stream>>>(s, d, n);
  };
  conv(gW1, W1g, 256 * 768); conv(cW1, W1c, 128 * 768); conv(dW1, W1d, 64 * 768);
  conv(gW2, W2g, 16 * 256);  conv(cW2, W2c, 16 * 128);  conv(dW2, W2d, 8 * 64);
  conv(mW1, mW1b, 384 * 768); conv(mW2, mW2b, 768 * 384);

  gather_kernel<<<dim3(256, BB), 256, 0, stream>>>(feat, coords, idx, cf, cc);
  safety_kernel<<<BB, 256, 0, stream>>>(cc, sg, sc, sd);

  score_kernel<<<dim3(16, BB), 256, 0, stream>>>(cf, sg, W1g, gb1, W2g, gb2, scg, 256, 16);
  score_kernel<<<dim3(8,  BB), 256, 0, stream>>>(cf, sc, W1c, cb1, W2c, cb2, scc, 128, 16);
  score_kernel<<<dim3(4,  BB), 256, 0, stream>>>(cf, sd, W1d, db1, W2d, db2, scd, 64, 8);

  topk_gather_kernel<<<BB, 256, 0, stream>>>(scg, scc, scd, cf, tok);

  mlp_kernel<<<10, 256, 0, stream>>>(tok, mW1b, mb1, mW2b, mb2, lng, lnb, out);
}
